// MultiHeadLatentAttention_79611513799085
// MI455X (gfx1250) — matrix operand order repaired and checked by reading, not compiled, not run on hardware
//
#include <hip/hip_runtime.h>
#include <hip/hip_bf16.h>

// ---------------- problem constants ----------------
#define NH     32          // heads
#define HD     128         // head dim (v / non-rope part)
#define QC     1536
#define KC     512
#define RD     64          // rope dim
#define DM     4096        // model dim
#define SEQ    2048
#define QKD    192         // HD + RD
#define HCOLS  2624        // QC + 2*KC + RD
#define QQRC   4160        // DM + RD
#define KVC    8192        // 2*DM
#define ATT_SCALE 0.07216878364870323f

typedef __attribute__((ext_vector_type(16))) __bf16 v16bf;
typedef __attribute__((ext_vector_type(8)))  float  v8f;
typedef int v4i_t __attribute__((vector_size(16)));

union V16 { v16bf v; unsigned short u[16]; };

__device__ __forceinline__ unsigned short f2bf(float f) {
  unsigned int x = __float_as_uint(f);
  x += 0x7FFFu + ((x >> 16) & 1u);            // round-to-nearest-even
  return (unsigned short)(x >> 16);
}

__device__ __forceinline__ v8f v8f_zero() {
  v8f z = {0.f,0.f,0.f,0.f,0.f,0.f,0.f,0.f};
  return z;
}

// 16-byte global->LDS copy: async DMA path on gfx1250, plain fallback.
// Builtin signature (from hipcc diagnostic): (as1 int4*, as3 int4*, imm, imm).
__device__ __forceinline__ void cp_g2l_16(const unsigned short* g,
                                          unsigned short* l) {
#if defined(__gfx1250__) && __has_builtin(__builtin_amdgcn_global_load_async_to_lds_b128)
  __builtin_amdgcn_global_load_async_to_lds_b128(
      (__attribute__((address_space(1))) v4i_t*)g,
      (__attribute__((address_space(3))) v4i_t*)l, 0, 0);
#else
  *(uint4*)l = *(const uint4*)g;
#endif
}

__device__ __forceinline__ void wait_async_copies() {
#if defined(__gfx1250__) && __has_builtin(__builtin_amdgcn_s_wait_asynccnt)
  __builtin_amdgcn_s_wait_asynccnt(0);
#endif
}

// =====================================================================
// fp32 -> bf16 elementwise (vectorized x4)
// =====================================================================
__global__ void __launch_bounds__(256)
convert_bf16(const float* __restrict__ src, unsigned short* __restrict__ dst,
             int n4)
{
  int i = blockIdx.x * 256 + threadIdx.x;
  if (i >= n4) return;
  float4 v = ((const float4*)src)[i];
  union { unsigned short u[4]; uint2 w; } o;
  o.u[0] = f2bf(v.x); o.u[1] = f2bf(v.y);
  o.u[2] = f2bf(v.z); o.u[3] = f2bf(v.w);
  ((uint2*)dst)[i] = o.w;
}

// =====================================================================
// fp32 [K][N] -> bf16 [N][K] tiled transpose-convert (weights, one pass)
// =====================================================================
__global__ void __launch_bounds__(256)
convert_transpose(const float* __restrict__ src, unsigned short* __restrict__ dst,
                  int K, int N)
{
  __shared__ unsigned short tile[32][33];
  const int n0 = blockIdx.x * 32, k0 = blockIdx.y * 32;
  const int tx = threadIdx.x & 31, ty = threadIdx.x >> 5;    // ty 0..7
  #pragma unroll
  for (int i = ty; i < 32; i += 8)
    tile[i][tx] = f2bf(src[(size_t)(k0 + i) * N + n0 + tx]); // tile[k][n]
  __syncthreads();
  #pragma unroll
  for (int i = ty; i < 32; i += 8)
    dst[(size_t)(n0 + i) * K + k0 + tx] = tile[tx][i];
}

// =====================================================================
// GEMM: C[M,N] f32 = A[M,K] bf16 (row-major, stride lda)
//                  * B[K,N]  given as Bt[N,K] bf16 (row-major, stride ldbt)
// BM=256, BN=64, BK=32; 256 threads = 8 waves (4M x 2N), wave tile 64x32.
// Double-buffered LDS, async global->LDS tile fills.
// =====================================================================
#define GBM 256
#define GBN 64
__global__ void __launch_bounds__(256)
gemm_bf16_wmma(const unsigned short* __restrict__ A, int lda,
               const unsigned short* __restrict__ Bt, int ldbt,
               float* __restrict__ C, int ldc,
               int M, int N, int K)
{
  __shared__ unsigned short As[2][GBM * 40];   // [m][k], row stride 80B
  __shared__ unsigned short Bs[2][GBN * 40];   // [n][k], row stride 80B

  const int tid  = threadIdx.x;
  const int wid  = tid >> 5;
  const int lane = tid & 31;
  const int half = lane >> 4;
  const int l16  = lane & 15;
  const int wm   = wid & 3;        // 4 wave rows  x64 = 256
  const int wn   = wid >> 2;       // 2 wave cols  x32 = 64
  const int bm   = blockIdx.y * GBM;
  const int bn   = blockIdx.x * GBN;

  v8f acc[4][2];
  #pragma unroll
  for (int i = 0; i < 4; ++i)
    #pragma unroll
    for (int j = 0; j < 2; ++j) acc[i][j] = v8f_zero();

  const int brow   = tid >> 2;     // 0..63
  const int bchunk = tid & 3;      // 0..3 (16B units)

  auto stage = [&](int buf, int kb) {
    // A tile: 256 rows x 64B -> 4 chunks per thread (thread == row)
    const unsigned short* ag = A + (size_t)(bm + tid) * lda + kb;
    unsigned short* al = &As[buf][tid * 40];
    #pragma unroll
    for (int j = 0; j < 4; ++j)
      cp_g2l_16(ag + j * 8, al + j * 8);
    // B tile: 64 rows x 64B -> 1 chunk per thread
    cp_g2l_16(Bt + (size_t)(bn + brow) * ldbt + kb + bchunk * 8,
              &Bs[buf][brow * 40 + bchunk * 8]);
  };

  const int nsteps = K >> 5;
  stage(0, 0);
  for (int s = 0; s < nsteps; ++s) {
    wait_async_copies();
    __syncthreads();                       // tile ready; prev buffer free
    const int cur = s & 1;
    if (s + 1 < nsteps) stage(cur ^ 1, (s + 1) << 5);

    V16 af[4], bf[2];
    #pragma unroll
    for (int mt = 0; mt < 4; ++mt) {
      const unsigned short* p = &As[cur][(wm * 64 + mt * 16 + l16) * 40];
      #pragma unroll
      for (int i = 0; i < 8; ++i) {
        af[mt].u[i]     = p[half * 16 + i];       // K 0-7  / 16-23
        af[mt].u[8 + i] = p[half * 16 + 8 + i];   // K 8-15 / 24-31
      }
    }
    #pragma unroll
    for (int nt = 0; nt < 2; ++nt) {
      const unsigned short* p =
          &Bs[cur][(wn * 32 + nt * 16 + l16) * 40 + half * 16];
      #pragma unroll
      for (int i = 0; i < 16; ++i) bf[nt].u[i] = p[i]; // K 0-15 / 16-31
    }
    #pragma unroll
    for (int mt = 0; mt < 4; ++mt)
      #pragma unroll
      for (int nt = 0; nt < 2; ++nt)
        acc[mt][nt] = __builtin_amdgcn_wmma_f32_16x16x32_bf16(
            false, af[mt].v, false, bf[nt].v, (short)0, acc[mt][nt],
            false, false);
  }

  #pragma unroll
  for (int mt = 0; mt < 4; ++mt) {
    #pragma unroll
    for (int nt = 0; nt < 2; ++nt) {
      const int col = bn + wn * 32 + nt * 16 + l16;
      #pragma unroll
      for (int r = 0; r < 8; ++r) {
        const int row = bm + wm * 64 + mt * 16 + half * 8 + r;
        C[(size_t)row * ldc + col] = acc[mt][nt][r];
      }
    }
  }
}

// =====================================================================
// Head packing + RoPE
// =====================================================================
__global__ void __launch_bounds__(256)
build_q_kernel(const float* __restrict__ qqr, const float* __restrict__ cs,
               const float* __restrict__ sn, unsigned short* __restrict__ Qh)
{
  int idx = blockIdx.x * 256 + threadIdx.x;      // [h][s][d], d 0..191
  if (idx >= NH * SEQ * QKD) return;
  int d = idx % QKD;
  int s = (idx / QKD) % SEQ;
  int h = idx / (QKD * SEQ);
  float val;
  if (d < HD) {
    val = qqr[(size_t)s * QQRC + h * HD + d];
  } else {
    int j = d - HD;
    float x = qqr[(size_t)s * QQRC + DM + j];
    float other = (j < RD / 2) ? -qqr[(size_t)s * QQRC + DM + j + RD / 2]
                               :  qqr[(size_t)s * QQRC + DM + j - RD / 2];
    val = x * cs[s * RD + j] + other * sn[s * RD + j];
  }
  Qh[idx] = f2bf(val);
}

__global__ void __launch_bounds__(256)
build_k_kernel(const float* __restrict__ kv, const float* __restrict__ hbuf,
               const float* __restrict__ cs, const float* __restrict__ sn,
               unsigned short* __restrict__ Kh)
{
  int idx = blockIdx.x * 256 + threadIdx.x;
  if (idx >= NH * SEQ * QKD) return;
  int d = idx % QKD;
  int s = (idx / QKD) % SEQ;
  int h = idx / (QKD * SEQ);
  float val;
  if (d < HD) {
    val = kv[(size_t)s * KVC + h * HD + d];
  } else {
    int j = d - HD;                               // k_rope = h[:, 2560+j]
    float x = hbuf[(size_t)s * HCOLS + QC + 2 * KC + j];
    float other = (j < RD / 2)
        ? -hbuf[(size_t)s * HCOLS + QC + 2 * KC + j + RD / 2]
        :  hbuf[(size_t)s * HCOLS + QC + 2 * KC + j - RD / 2];
    val = x * cs[s * RD + j] + other * sn[s * RD + j];
  }
  Kh[idx] = f2bf(val);
}

// V transposed per head: VhT[h][d][s]  (so attention V-tiles are contiguous)
__global__ void __launch_bounds__(256)
build_vT_kernel(const float* __restrict__ kv, unsigned short* __restrict__ VhT)
{
  __shared__ unsigned short tile[32][33];
  const int h  = blockIdx.z;
  const int d0 = blockIdx.y * 32;
  const int s0 = blockIdx.x * 32;
  const int tx = threadIdx.x & 31, ty = threadIdx.x >> 5;
  #pragma unroll
  for (int i = ty; i < 32; i += 8)
    tile[i][tx] = f2bf(kv[(size_t)(s0 + i) * KVC + DM + h * HD + d0 + tx]);
  __syncthreads();
  #pragma unroll
  for (int i = ty; i < 32; i += 8)
    VhT[(size_t)(h * HD + d0 + i) * SEQ + s0 + tx] = tile[tx][i];
}

// =====================================================================
// Causal flash attention. Block = (head, 64 q rows), 4 waves x 16 rows.
// Per 32-key step: cooperative async staging of K-tile (32x192) and
// transposed V-tile (128x32) in LDS; 12 WMMAs S=Q*K^T, online softmax,
// LDS bounce for P (C->A layout), 8 WMMAs O+=P*V. Output bf16.
// =====================================================================
__global__ void __launch_bounds__(128)
mla_attention_kernel(const unsigned short* __restrict__ Qh,
                     const unsigned short* __restrict__ Kh,
                     const unsigned short* __restrict__ VhT,
                     unsigned short* __restrict__ attn)
{
  __shared__ unsigned short Kt[32 * 200];   // [key][d], row stride 400B
  __shared__ unsigned short Vt[128 * 40];   // [d][key], row stride 80B
  __shared__ unsigned short sP[4 * 512];    // 1KB per wave

  const int tid  = threadIdx.x;
  const int wid  = tid >> 5;
  const int lane = tid & 31;
  const int half = lane >> 4;
  const int l16  = lane & 15;
  const int head = blockIdx.y;
  const int qb   = blockIdx.x * 64;
  const int q_base = qb + wid * 16;
  unsigned short* P = sP + wid * 512;

  const unsigned short* Qp = Qh  + (size_t)head * SEQ * QKD;
  const unsigned short* Kg = Kh  + (size_t)head * SEQ * QKD;
  const unsigned short* Vg = VhT + (size_t)head * HD * SEQ;

  // Q A-fragments, full dq=192 (6 x K32), kept in registers
  V16 qf[6];
  {
    const unsigned short* qrow = Qp + (size_t)(q_base + l16) * QKD;
    #pragma unroll
    for (int t = 0; t < 6; ++t) {
      const int k0 = t * 32;
      #pragma unroll
      for (int i = 0; i < 8; ++i) {
        qf[t].u[i]     = qrow[k0 + half * 16 + i];
        qf[t].u[8 + i] = qrow[k0 + half * 16 + 8 + i];
      }
    }
  }

  v8f oacc[8];
  #pragma unroll
  for (int n = 0; n < 8; ++n) oacc[n] = v8f_zero();
  float m_i[8], l_i[8];
  #pragma unroll
  for (int r = 0; r < 8; ++r) { m_i[r] = -1e30f; l_i[r] = 0.f; }

  const int kend = qb + 63;                    // causal bound for the block
  for (int kb = 0; kb <= kend; kb += 32) {
    // ---- stage K tile: 32 rows x 384B = 768 chunks / 128 threads ----
    #pragma unroll
    for (int c = 0; c < 6; ++c) {
      const int chunk = tid + c * 128;
      const int row = chunk / 24, col = chunk % 24;
      cp_g2l_16(Kg + (size_t)(kb + row) * QKD + col * 8,
                Kt + row * 200 + col * 8);
    }
    // ---- stage V tile: 128 rows x 64B = 512 chunks (thread == d row) ----
    #pragma unroll
    for (int j = 0; j < 4; ++j)
      cp_g2l_16(Vg + (size_t)tid * SEQ + kb + j * 8,
                Vt + tid * 40 + j * 8);
    wait_async_copies();
    __syncthreads();

    // ---- S tiles: keys kb..kb+15 and kb+16..kb+31 ----
    v8f s0 = v8f_zero(), s1 = v8f_zero();
    #pragma unroll
    for (int t = 0; t < 6; ++t) {
      const int k0 = t * 32;
      V16 kf0, kf1;
      const unsigned short* p0 = Kt + l16 * 200 + k0 + half * 16;
      const unsigned short* p1 = Kt + (16 + l16) * 200 + k0 + half * 16;
      #pragma unroll
      for (int i = 0; i < 16; ++i) { kf0.u[i] = p0[i]; kf1.u[i] = p1[i]; }
      s0 = __builtin_amdgcn_wmma_f32_16x16x32_bf16(false, qf[t].v, false, kf0.v,
                                                   (short)0, s0, false, false);
      s1 = __builtin_amdgcn_wmma_f32_16x16x32_bf16(false, qf[t].v, false, kf1.v,
                                                   (short)0, s1, false, false);
    }

    // ---- online softmax ----
    #pragma unroll
    for (int r = 0; r < 8; ++r) {
      const int qrow = q_base + half * 8 + r;
      const int key0 = kb + l16;
      float v0 = s0[r] * ATT_SCALE; if (key0 > qrow)      v0 = -1e30f;
      float v1 = s1[r] * ATT_SCALE; if (key0 + 16 > qrow) v1 = -1e30f;
      float mx = fmaxf(v0, v1);
      #pragma unroll
      for (int off = 1; off < 16; off <<= 1)
        mx = fmaxf(mx, __shfl_xor(mx, off, 32));   // stays in 16-lane half
      const float mn = fmaxf(m_i[r], mx);
      const float f  = __expf(m_i[r] - mn);
      const float p0 = __expf(v0 - mn);
      const float p1 = __expf(v1 - mn);
      float rs = p0 + p1;
      #pragma unroll
      for (int off = 1; off < 16; off <<= 1)
        rs += __shfl_xor(rs, off, 32);
      l_i[r] = l_i[r] * f + rs;
      m_i[r] = mn;
      #pragma unroll
      for (int n = 0; n < 8; ++n) oacc[n][r] *= f;
      P[(half * 8 + r) * 32 + l16]      = f2bf(p0);
      P[(half * 8 + r) * 32 + 16 + l16] = f2bf(p1);
    }

    // ---- P as 16x32 A-fragment ----
    V16 pf;
    {
      const unsigned short* prow = P + l16 * 32;
      #pragma unroll
      for (int i = 0; i < 8; ++i) {
        pf.u[i]     = prow[half * 16 + i];
        pf.u[8 + i] = prow[half * 16 + 8 + i];
      }
    }

    // ---- O += P * V ----
    #pragma unroll
    for (int n = 0; n < 8; ++n) {
      V16 vf;
      const unsigned short* p = Vt + (n * 16 + l16) * 40 + half * 16;
      #pragma unroll
      for (int i = 0; i < 16; ++i) vf.u[i] = p[i];
      oacc[n] = __builtin_amdgcn_wmma_f32_16x16x32_bf16(false, pf.v, false, vf.v,
                                                        (short)0, oacc[n],
                                                        false, false);
    }
    __syncthreads();                 // tiles free for next staging
  }

  // ---- normalize, write bf16 [s][h*128 + d] ----
  #pragma unroll
  for (int r = 0; r < 8; ++r) {
    const int row = q_base + half * 8 + r;
    const float inv = 1.f / l_i[r];
    #pragma unroll
    for (int n = 0; n < 8; ++n)
      attn[(size_t)row * DM + head * HD + n * 16 + l16] =
          f2bf(oacc[n][r] * inv);
  }
}

// =====================================================================
// Host launcher
// =====================================================================
extern "C" void kernel_launch(void* const* d_in, const int* in_sizes, int n_in,
                              void* d_out, int out_size, void* d_ws, size_t ws_size,
                              hipStream_t stream) {
  (void)in_sizes; (void)n_in; (void)out_size; (void)ws_size;

  const float* hs      = (const float*)d_in[0];   // [2048,4096]
  const float* cosb    = (const float*)d_in[1];   // [2048,64]
  const float* sinb    = (const float*)d_in[2];   // [2048,64]
  const float* w_down  = (const float*)d_in[3];   // [4096,2624]
  const float* w_q_up  = (const float*)d_in[4];   // [1536,4160]
  const float* w_kv_up = (const float*)d_in[5];   // [1024,8192]
  const float* w_proj  = (const float*)d_in[6];   // [4096,4096]
  float* out = (float*)d_out;                     // [2048,4096] f32

  char* ws = (char*)d_ws;
  size_t off = 0;
  auto take = [&](size_t bytes) {
    char* p = ws + off;
    off += (bytes + 255) & ~(size_t)255;
    return p;
  };
  float* h_f32   = (float*)take((size_t)SEQ * HCOLS * 4);
  float* qqr_f32 = (float*)take((size_t)SEQ * QQRC * 4);
  float* kv_f32  = (float*)take((size_t)SEQ * KVC * 4);
  unsigned short* hs_bf  = (unsigned short*)take((size_t)SEQ * DM * 2);
  unsigned short* h_bf   = (unsigned short*)take((size_t)SEQ * HCOLS * 2);
  unsigned short* attn_bf= (unsigned short*)take((size_t)SEQ * DM * 2);
  unsigned short* wdT  = (unsigned short*)take((size_t)DM * HCOLS * 2);
  unsigned short* wqT  = (unsigned short*)take((size_t)QC * QQRC * 2);
  unsigned short* wkvT = (unsigned short*)take((size_t)2 * KC * KVC * 2);
  unsigned short* wpT  = (unsigned short*)take((size_t)DM * DM * 2);
  unsigned short* Qhb  = (unsigned short*)take((size_t)NH * SEQ * QKD * 2);
  unsigned short* Khb  = (unsigned short*)take((size_t)NH * SEQ * QKD * 2);
  unsigned short* VhTb = (unsigned short*)take((size_t)NH * HD * SEQ * 2);

  const dim3 blk(256);

  // one-time precision/layout conversion
  int n4 = SEQ * DM / 4;
  convert_bf16<<<(n4 + 255) / 256, blk, 0, stream>>>(hs, hs_bf, n4);
  convert_transpose<<<dim3(HCOLS / 32, DM / 32), blk, 0, stream>>>(
      w_down, wdT, DM, HCOLS);
  convert_transpose<<<dim3(QQRC / 32, QC / 32), blk, 0, stream>>>(
      w_q_up, wqT, QC, QQRC);
  convert_transpose<<<dim3(KVC / 32, (2 * KC) / 32), blk, 0, stream>>>(
      w_kv_up, wkvT, 2 * KC, KVC);
  convert_transpose<<<dim3(DM / 32, DM / 32), blk, 0, stream>>>(
      w_proj, wpT, DM, DM);

  // h = hs @ w_down
  gemm_bf16_wmma<<<dim3(HCOLS / GBN, SEQ / GBM), blk, 0, stream>>>(
      hs_bf, DM, wdT, DM, h_f32, HCOLS, SEQ, HCOLS, DM);
  n4 = SEQ * HCOLS / 4;
  convert_bf16<<<(n4 + 255) / 256, blk, 0, stream>>>(h_f32, h_bf, n4);

  // qqr = cq @ w_q_up ; kv = ckv @ w_kv_up
  gemm_bf16_wmma<<<dim3(QQRC / GBN, SEQ / GBM), blk, 0, stream>>>(
      h_bf, HCOLS, wqT, QC, qqr_f32, QQRC, SEQ, QQRC, QC);
  gemm_bf16_wmma<<<dim3(KVC / GBN, SEQ / GBM), blk, 0, stream>>>(
      h_bf + QC, HCOLS, wkvT, 2 * KC, kv_f32, KVC, SEQ, KVC, 2 * KC);

  // pack heads (+ RoPE)
  const int nq = NH * SEQ * QKD;
  build_q_kernel<<<(nq + 255) / 256, blk, 0, stream>>>(qqr_f32, cosb, sinb, Qhb);
  build_k_kernel<<<(nq + 255) / 256, blk, 0, stream>>>(kv_f32, h_f32, cosb, sinb, Khb);
  build_vT_kernel<<<dim3(SEQ / 32, HD / 32, NH), blk, 0, stream>>>(kv_f32, VhTb);

  // causal flash attention -> bf16
  mla_attention_kernel<<<dim3(SEQ / 64, NH), dim3(128), 0, stream>>>(
      Qhb, Khb, VhTb, attn_bf);

  // out = attn @ w_proj
  gemm_bf16_wmma<<<dim3(DM / GBN, SEQ / GBM), blk, 0, stream>>>(
      attn_bf, DM, wpT, DM, out, DM, SEQ, DM, DM);
}
